// Sliding_attention_Network_75307956568151
// MI455X (gfx1250) — matrix operand order repaired and checked by reading, not compiled, not run on hardware
//
#include <hip/hip_runtime.h>
#include <math.h>

// ---------------------------------------------------------------------------
// CDNA5 (gfx1250) sliding-attention network forward pass.
// GEMMs: v_wmma_f32_16x16x32_bf16, 64x64 macro tile, 4 waves x 4 WMMA tiles,
// A staged via global_load_async_to_lds_b128 (ASYNCcnt), fragments fetched
// with contiguous ds_load_b128 per ISA 7.12.2 16-bit layouts.
// ---------------------------------------------------------------------------

typedef __bf16 bf16_t;
typedef bf16_t v16bf __attribute__((ext_vector_type(16)));
typedef bf16_t v8bf  __attribute__((ext_vector_type(8)));
typedef bf16_t v2bf  __attribute__((ext_vector_type(2)));
typedef float  v8f   __attribute__((ext_vector_type(8)));

#define LN_EPS 1e-5f

__device__ __forceinline__ float gelu_erf(float x) {
  return 0.5f * x * (1.0f + erff(x * 0.70710678118654752440f));
}

// ---------------------------------------------------------------------------
// im2col: x [B,C,H,W] -> cols [B, Ho*Wo, C*kh*kw]   ((c,i,j)-major, c outer)
// ---------------------------------------------------------------------------
__global__ void im2col_k(const float* __restrict__ x, float* __restrict__ cols,
                         int B, int C, int H, int W,
                         int kh, int kw, int s, int pad, int Ho, int Wo) {
  const int K = C * kh * kw;
  const long long total = (long long)B * Ho * Wo * K;
  for (long long idx = blockIdx.x * (long long)blockDim.x + threadIdx.x;
       idx < total; idx += (long long)gridDim.x * blockDim.x) {
    int k = (int)(idx % K);
    long long t = idx / K;
    int n = (int)(t % (Ho * Wo));
    int b = (int)(t / (Ho * Wo));
    int c = k / (kh * kw);
    int r = k % (kh * kw);
    int i = r / kw, j = r % kw;
    int oh = n / Wo, ow = n % Wo;
    int ih = oh * s + i - pad;
    int iw = ow * s + j - pad;
    float v = 0.0f;
    if (ih >= 0 && ih < H && iw >= 0 && iw < W)
      v = x[(((long long)b * C + c) * H + ih) * W + iw];
    cols[idx] = v;
  }
}

// ---------------------------------------------------------------------------
// Batched/strided GEMM:  D[b] = A[b](MxK) * B[b](KxN) (+bias), row-major.
//   macro tile 64x64, 128 threads = 4 waves (2x2), each wave 32x32 = 4 WMMA.
//   K stepped by 32.  A tile: async-copied raw fp32 into LDS (interior path),
//   then packed to bf16.  B tile staged [n][k]-major so fragments are
//   contiguous.  transB: B element (k,n) read as Bm[n*ldb + k].
// ---------------------------------------------------------------------------
#define TMB 64
#define TNB 64
#define TKB 32

union Frag { v16bf v; v8bf h[2]; };

__global__ __launch_bounds__(128)
void gemm_wmma_k(const float* __restrict__ A, const float* __restrict__ Bm,
                 const float* __restrict__ bias, float* __restrict__ D,
                 int M, int N, int K, int ldb, int transB,
                 long long sA, long long sB, long long sD) {
  __shared__ float  rawA[TMB * TKB];   // 8 KB raw fp32 A tile (async target)
  __shared__ bf16_t aS[TMB * TKB];     // 4 KB bf16 A tile, [m][k]
  __shared__ bf16_t bS[TNB * TKB];     // 4 KB bf16 B tile, [n][k]

  const int bz = blockIdx.z;
  A  += (long long)bz * sA;
  Bm += (long long)bz * sB;
  D  += (long long)bz * sD;
  const int tm0 = blockIdx.y * TMB;
  const int tn0 = blockIdx.x * TNB;
  const int tid  = threadIdx.x;
  const int lane = tid & 31;
  const int wave = tid >> 5;
  const int wm = (wave & 1) * 32;
  const int wn = (wave >> 1) * 32;
  const int hi  = (lane >> 4) & 1;
  const int l15 = lane & 15;

  v8f acc00 = {}, acc01 = {}, acc10 = {}, acc11 = {};

  for (int k0 = 0; k0 < K; k0 += TKB) {
    const bool interior = (tm0 + TMB <= M) && (tn0 + TNB <= N) && (k0 + TKB <= K);

    if (interior) {
      // ---- A tile: per-lane async 16B copies, global -> LDS (ASYNCcnt) ----
      // 64 rows x 32 floats = 512 float4 chunks; 128 threads -> 4 each.
      for (int q = tid; q < (TMB * TKB) / 4; q += 128) {
        int row = q >> 3;            // 8 float4 per row
        int c4  = (q & 7) * 4;
        const float* gp = A + (long long)(tm0 + row) * K + (k0 + c4);
        unsigned lofs = (unsigned)(size_t)&rawA[row * TKB + c4];
        asm volatile("global_load_async_to_lds_b128 %0, %1, off"
                     :: "v"(lofs), "v"((unsigned long long)(size_t)gp)
                     : "memory");
      }
      // prefetch next B tile rows while async copy flies
      if (k0 + TKB < K) {
        int pn = tn0 + (tid & (TNB - 1));
        const float* pb = transB ? &Bm[(long long)pn * ldb + k0 + TKB]
                                 : &Bm[(long long)(k0 + TKB) * ldb + pn];
        __builtin_prefetch(pb, 0, 0);
      }
      // ---- B tile -> bS[n][k] (pairs, packed 32-bit stores) ----
      for (int q = tid; q < (TNB * TKB) / 2; q += 128) {
        int n = q >> 4, kp = (q & 15) * 2;
        int gn = tn0 + n, gk = k0 + kp;
        float f0, f1;
        if (transB) {
          const float* p = Bm + (long long)gn * ldb + gk;
          f0 = p[0]; f1 = p[1];
        } else {
          f0 = Bm[(long long)gk * ldb + gn];
          f1 = Bm[(long long)(gk + 1) * ldb + gn];
        }
        v2bf pk; pk[0] = (bf16_t)f0; pk[1] = (bf16_t)f1;
        *(v2bf*)&bS[n * TKB + kp] = pk;
      }
      asm volatile("s_wait_asynccnt 0x0" ::: "memory");
    } else {
      // ---- boundary path: guarded scalar staging ----
      for (int q = tid; q < TMB * TKB; q += 128) {
        int row = q >> 5, c = q & 31;
        int gm = tm0 + row, gk = k0 + c;
        rawA[q] = (gm < M && gk < K) ? A[(long long)gm * K + gk] : 0.0f;
      }
      for (int q = tid; q < (TNB * TKB) / 2; q += 128) {
        int n = q >> 4, kp = (q & 15) * 2;
        int gn = tn0 + n, gk = k0 + kp;
        float f0 = 0.0f, f1 = 0.0f;
        if (gn < N) {
          if (transB) {
            if (gk     < K) f0 = Bm[(long long)gn * ldb + gk];
            if (gk + 1 < K) f1 = Bm[(long long)gn * ldb + gk + 1];
          } else {
            if (gk     < K) f0 = Bm[(long long)gk * ldb + gn];
            if (gk + 1 < K) f1 = Bm[(long long)(gk + 1) * ldb + gn];
          }
        }
        v2bf pk; pk[0] = (bf16_t)f0; pk[1] = (bf16_t)f1;
        *(v2bf*)&bS[n * TKB + kp] = pk;
      }
    }
    __syncthreads();

    // ---- pack rawA (fp32) -> aS (bf16), 32-bit stores ----
    for (int q = tid; q < (TMB * TKB) / 2; q += 128) {
      int row = q >> 4, kp = (q & 15) * 2;
      v2bf pk;
      pk[0] = (bf16_t)rawA[row * TKB + kp];
      pk[1] = (bf16_t)rawA[row * TKB + kp + 1];
      *(v2bf*)&aS[row * TKB + kp] = pk;
    }
    __syncthreads();

    // ---- fragments (contiguous b128 loads per ISA 7.12.2) ----
    Frag a0, a1, b0, b1;
    const int ra0 = (wm + l15) * TKB;
    const int ra1 = (wm + 16 + l15) * TKB;
    a0.h[0] = *(const v8bf*)&aS[ra0 + hi * 8];
    a0.h[1] = *(const v8bf*)&aS[ra0 + 16 + hi * 8];
    a1.h[0] = *(const v8bf*)&aS[ra1 + hi * 8];
    a1.h[1] = *(const v8bf*)&aS[ra1 + 16 + hi * 8];
    const int rb0 = (wn + l15) * TKB;
    const int rb1 = (wn + 16 + l15) * TKB;
    b0.h[0] = *(const v8bf*)&bS[rb0 + hi * 8];
    b0.h[1] = *(const v8bf*)&bS[rb0 + 16 + hi * 8];
    b1.h[0] = *(const v8bf*)&bS[rb1 + hi * 8];
    b1.h[1] = *(const v8bf*)&bS[rb1 + 16 + hi * 8];

    acc00 = __builtin_amdgcn_wmma_f32_16x16x32_bf16(false, a0.v, false, b0.v,
                                                    (short)0, acc00, false, false);
    acc01 = __builtin_amdgcn_wmma_f32_16x16x32_bf16(false, a0.v, false, b1.v,
                                                    (short)0, acc01, false, false);
    acc10 = __builtin_amdgcn_wmma_f32_16x16x32_bf16(false, a1.v, false, b0.v,
                                                    (short)0, acc10, false, false);
    acc11 = __builtin_amdgcn_wmma_f32_16x16x32_bf16(false, a1.v, false, b1.v,
                                                    (short)0, acc11, false, false);
    __syncthreads();
  }

  // ---- store: C/D layout VGPR i -> M = +hi*8+i, N = +l15 ----
  v8f accs[2][2] = {{acc00, acc01}, {acc10, acc11}};
#pragma unroll
  for (int ti = 0; ti < 2; ++ti) {
#pragma unroll
    for (int tj = 0; tj < 2; ++tj) {
      int nOut  = tn0 + wn + tj * 16 + l15;
      int mBase = tm0 + wm + ti * 16 + hi * 8;
#pragma unroll
      for (int i = 0; i < 8; ++i) {
        int mOut = mBase + i;
        if (mOut < M && nOut < N) {
          float v = accs[ti][tj][i];
          if (bias) v += bias[nOut];
          D[(long long)mOut * N + nOut] = v;
        }
      }
    }
  }
}

// ---------------------------------------------------------------------------
// LayerNorm over last dim, one block per row, optional fused GELU (in-place ok)
// ---------------------------------------------------------------------------
__global__ __launch_bounds__(256)
void layernorm_k(const float* __restrict__ x, const float* __restrict__ g,
                 const float* __restrict__ b, float* __restrict__ y,
                 int R, int C, int fuse_gelu) {
  __shared__ float red[256];
  for (int r = blockIdx.x; r < R; r += gridDim.x) {
    const float* xr = x + (long long)r * C;
    float s = 0.0f;
    for (int c = threadIdx.x; c < C; c += 256) s += xr[c];
    red[threadIdx.x] = s; __syncthreads();
    for (int o = 128; o > 0; o >>= 1) {
      if (threadIdx.x < o) red[threadIdx.x] += red[threadIdx.x + o];
      __syncthreads();
    }
    float mean = red[0] / (float)C; __syncthreads();
    float vs = 0.0f;
    for (int c = threadIdx.x; c < C; c += 256) { float d = xr[c] - mean; vs += d * d; }
    red[threadIdx.x] = vs; __syncthreads();
    for (int o = 128; o > 0; o >>= 1) {
      if (threadIdx.x < o) red[threadIdx.x] += red[threadIdx.x + o];
      __syncthreads();
    }
    float rstd = rsqrtf(red[0] / (float)C + LN_EPS); __syncthreads();
    float* yr = y + (long long)r * C;
    for (int c = threadIdx.x; c < C; c += 256) {
      float v = (xr[c] - mean) * rstd * g[c] + b[c];
      yr[c] = fuse_gelu ? gelu_erf(v) : v;
    }
    __syncthreads();
  }
}

// ---------------------------------------------------------------------------
// Row softmax with pre-scale (in place)
// ---------------------------------------------------------------------------
__global__ __launch_bounds__(256)
void softmax_k(float* __restrict__ x, int R, int C, float scale) {
  __shared__ float red[256];
  for (int r = blockIdx.x; r < R; r += gridDim.x) {
    float* xr = x + (long long)r * C;
    float mx = -3.4e38f;
    for (int c = threadIdx.x; c < C; c += 256) mx = fmaxf(mx, xr[c] * scale);
    red[threadIdx.x] = mx; __syncthreads();
    for (int o = 128; o > 0; o >>= 1) {
      if (threadIdx.x < o) red[threadIdx.x] = fmaxf(red[threadIdx.x], red[threadIdx.x + o]);
      __syncthreads();
    }
    mx = red[0]; __syncthreads();
    float s = 0.0f;
    for (int c = threadIdx.x; c < C; c += 256) {
      float e = __expf(xr[c] * scale - mx);
      xr[c] = e; s += e;
    }
    red[threadIdx.x] = s; __syncthreads();
    for (int o = 128; o > 0; o >>= 1) {
      if (threadIdx.x < o) red[threadIdx.x] += red[threadIdx.x + o];
      __syncthreads();
    }
    float inv = 1.0f / red[0]; __syncthreads();
    for (int c = threadIdx.x; c < C; c += 256) xr[c] *= inv;
    __syncthreads();
  }
}

// ---------------------------------------------------------------------------
// misc elementwise / reductions
// ---------------------------------------------------------------------------
__global__ void gelu_k(float* __restrict__ x, long long n) {
  for (long long i = blockIdx.x * (long long)blockDim.x + threadIdx.x; i < n;
       i += (long long)gridDim.x * blockDim.x)
    x[i] = gelu_erf(x[i]);
}

__global__ void add_k(const float* __restrict__ a, const float* __restrict__ b,
                      float* __restrict__ y, long long n) {
  for (long long i = blockIdx.x * (long long)blockDim.x + threadIdx.x; i < n;
       i += (long long)gridDim.x * blockDim.x)
    y[i] = a[i] + b[i];
}

// x [B,R,C] -> y [B,C,R]
__global__ void transpose_k(const float* __restrict__ x, float* __restrict__ y,
                            int B, int R, int C) {
  long long total = (long long)B * R * C;
  for (long long idx = blockIdx.x * (long long)blockDim.x + threadIdx.x; idx < total;
       idx += (long long)gridDim.x * blockDim.x) {
    int c = (int)(idx % C);
    long long t = idx / C;
    int r = (int)(t % R);
    int b = (int)(t / R);
    y[((long long)b * C + c) * R + r] = x[idx];
  }
}

// channel gate: t[b,c,:] *= (dot(t[b,c,:], Wc) + bc)   — t is [BC, N] rows
__global__ __launch_bounds__(256)
void gate_k(float* __restrict__ t, const float* __restrict__ Wc,
            const float* __restrict__ bc, int BC, int N) {
  __shared__ float red[256];
  for (int rc = blockIdx.x; rc < BC; rc += gridDim.x) {
    float* tr = t + (long long)rc * N;
    float s = 0.0f;
    for (int n = threadIdx.x; n < N; n += 256) s += tr[n] * Wc[n];
    red[threadIdx.x] = s; __syncthreads();
    for (int o = 128; o > 0; o >>= 1) {
      if (threadIdx.x < o) red[threadIdx.x] += red[threadIdx.x + o];
      __syncthreads();
    }
    float gate = red[0] + bc[0]; __syncthreads();
    for (int n = threadIdx.x; n < N; n += 256) tr[n] *= gate;
    __syncthreads();
  }
}

// BatchNorm stats: x [B,C,N], per-channel sum & sumsq over B*N
__global__ __launch_bounds__(256)
void bn_stats_k(const float* __restrict__ x, float* __restrict__ sum,
                float* __restrict__ sq, int B, int C, int N) {
  __shared__ float r1[256];
  __shared__ float r2[256];
  int c = blockIdx.x;
  float s = 0.0f, ss = 0.0f;
  int cnt = B * N;
  for (int i = threadIdx.x; i < cnt; i += 256) {
    int b = i / N, n = i % N;
    float v = x[((long long)b * C + c) * N + n];
    s += v; ss += v * v;
  }
  r1[threadIdx.x] = s; r2[threadIdx.x] = ss; __syncthreads();
  for (int o = 128; o > 0; o >>= 1) {
    if (threadIdx.x < o) {
      r1[threadIdx.x] += r1[threadIdx.x + o];
      r2[threadIdx.x] += r2[threadIdx.x + o];
    }
    __syncthreads();
  }
  if (threadIdx.x == 0) { sum[c] = r1[0]; sq[c] = r2[0]; }
}

__global__ void bn_apply_k(const float* __restrict__ x, const float* __restrict__ sum,
                           const float* __restrict__ sq, const float* __restrict__ g,
                           const float* __restrict__ b, float* __restrict__ y,
                           int B, int C, int N, int fuse_gelu) {
  long long total = (long long)B * C * N;
  float invcnt = 1.0f / (float)(B * N);
  for (long long idx = blockIdx.x * (long long)blockDim.x + threadIdx.x; idx < total;
       idx += (long long)gridDim.x * blockDim.x) {
    int c = (int)((idx / N) % C);
    float m = sum[c] * invcnt;
    float v = sq[c] * invcnt - m * m;
    float r = rsqrtf(v + LN_EPS);
    float o = (x[idx] - m) * r * g[c] + b[c];
    y[idx] = fuse_gelu ? gelu_erf(o) : o;
  }
}

// x [B,N,C] -> y [B,C]  (mean over N)
__global__ void mean_mid_k(const float* __restrict__ x, float* __restrict__ y,
                           int B, int N, int C) {
  long long total = (long long)B * C;
  for (long long idx = blockIdx.x * (long long)blockDim.x + threadIdx.x; idx < total;
       idx += (long long)gridDim.x * blockDim.x) {
    int c = (int)(idx % C);
    int b = (int)(idx / C);
    float s = 0.0f;
    for (int n = 0; n < N; ++n) s += x[((long long)b * N + n) * C + c];
    y[idx] = s / (float)N;
  }
}

// ---------------------------------------------------------------------------
// Host orchestration
// ---------------------------------------------------------------------------
static inline int ceil_div_i(int a, int b) { return (a + b - 1) / b; }
static inline int imin_i(int a, int b) { return a < b ? a : b; }
static inline int grid1d(long long n, int bs) {
  long long g = (n + bs - 1) / bs;
  if (g > 32768) g = 32768;
  if (g < 1) g = 1;
  return (int)g;
}

// sa-block param leaves in JAX sorted-key order
struct SAP {
  const float *Wc, *Wk, *Wq, *Wv, *bc, *be1, *be2, *bek, *beq, *bev,
              *bk, *bq, *bv, *g1, *g2, *gk, *gq, *gv;
};
static SAP load_sap(void* const* d, int base) {
  SAP s;
  const float** f = reinterpret_cast<const float**>(&s);
  for (int i = 0; i < 18; ++i) f[i] = reinterpret_cast<const float*>(d[base + i]);
  return s;
}

struct Scratch {
  float *cols, *q, *k, *v, *scores, *ctx, *t, *sum, *sq;
};

static void gemm(hipStream_t st, const float* A, const float* B, const float* bias,
                 float* D, int M, int N, int K, int ldb, int transB, int batch,
                 long long sA, long long sB, long long sD) {
  dim3 g(ceil_div_i(N, TNB), ceil_div_i(M, TMB), batch);
  gemm_wmma_k<<<g, dim3(128), 0, st>>>(A, B, bias, D, M, N, K, ldb, transB, sA, sB, sD);
}

static void sa_block(hipStream_t st, const SAP& P, const float* x, float* out,
                     const Scratch& S, int B, int inC, int H, int W,
                     int outC, int kh, int kw, int s, int pad) {
  const int Ho = (H + 2 * pad - kh) / s + 1;
  const int Wo = (W + 2 * pad - kw) / s + 1;
  const int N = Ho * Wo;
  const int K = inC * kh * kw;
  const int M = B * N;

  im2col_k<<<grid1d((long long)M * K, 256), 256, 0, st>>>(
      x, S.cols, B, inC, H, W, kh, kw, s, pad, Ho, Wo);

  // q/k/v projections (+bias) and per-row LayerNorm
  gemm(st, S.cols, P.Wq, P.bq, S.q, M, outC, K, outC, 0, 1, 0, 0, 0);
  layernorm_k<<<imin_i(M, 32768), 256, 0, st>>>(S.q, P.gq, P.beq, S.q, M, outC, 0);
  gemm(st, S.cols, P.Wk, P.bk, S.k, M, outC, K, outC, 0, 1, 0, 0, 0);
  layernorm_k<<<imin_i(M, 32768), 256, 0, st>>>(S.k, P.gk, P.bek, S.k, M, outC, 0);
  gemm(st, S.cols, P.Wv, P.bv, S.v, M, outC, K, outC, 0, 1, 0, 0, 0);
  layernorm_k<<<imin_i(M, 32768), 256, 0, st>>>(S.v, P.gv, P.bev, S.v, M, outC, 0);

  // scores = q @ k^T  (batched over B), softmax(scores / sqrt(C))
  gemm(st, S.q, S.k, nullptr, S.scores, N, N, outC, outC, 1, B,
       (long long)N * outC, (long long)N * outC, (long long)N * N);
  softmax_k<<<imin_i(M, 32768), 256, 0, st>>>(S.scores, M, N, 1.0f / sqrtf((float)outC));

  // ctx = attn @ v, then LN + GELU
  gemm(st, S.scores, S.v, nullptr, S.ctx, N, outC, N, outC, 0, B,
       (long long)N * N, (long long)N * outC, (long long)N * outC);
  layernorm_k<<<imin_i(M, 32768), 256, 0, st>>>(S.ctx, P.g1, P.be1, S.ctx, M, outC, 1);

  // [B,N,C] -> [B,C,N], channel-attention gate, BN(+GELU)
  transpose_k<<<grid1d((long long)M * outC, 256), 256, 0, st>>>(S.ctx, S.t, B, N, outC);
  gate_k<<<imin_i(B * outC, 32768), 256, 0, st>>>(S.t, P.Wc, P.bc, B * outC, N);
  bn_stats_k<<<outC, 256, 0, st>>>(S.t, S.sum, S.sq, B, outC, N);
  bn_apply_k<<<grid1d((long long)B * outC * N, 256), 256, 0, st>>>(
      S.t, S.sum, S.sq, P.g2, P.be2, out, B, outC, N, 1);
}

static void bn_merge(hipStream_t st, const Scratch& S, float* m, const float* sc,
                     float* out, const float* g, const float* b, int B, int C, int N) {
  add_k<<<grid1d((long long)B * C * N, 256), 256, 0, st>>>(m, sc, m, (long long)B * C * N);
  bn_stats_k<<<C, 256, 0, st>>>(m, S.sum, S.sq, B, C, N);
  bn_apply_k<<<grid1d((long long)B * C * N, 256), 256, 0, st>>>(
      m, S.sum, S.sq, g, b, out, B, C, N, 0);
}

extern "C" void kernel_launch(void* const* d_in, const int* in_sizes, int n_in,
                              void* d_out, int out_size, void* d_ws, size_t ws_size,
                              hipStream_t stream) {
  (void)in_sizes; (void)n_in; (void)out_size; (void)ws_size;
  const int B = 32;

  // JAX sorted-key DFS flattening of the params dict:
  // 0:x  1:W1 2:W2 3:Wp 4:bp 5:ffn.b 6:ffn.g 7:n2.b 8:n2.g 9:n3.b 10:n3.g
  // 11:n4.b 12:n4.g 13:proj.b 14:proj.g, then 18-leaf sa blocks:
  // s1,s2a,s2b,s2c,s3a,s3b,s3c,s4a,s4b,s4c,sc2,sc3,sc4
  const float* x      = (const float*)d_in[0];
  const float* W1     = (const float*)d_in[1];
  const float* W2     = (const float*)d_in[2];
  const float* Wp     = (const float*)d_in[3];
  const float* bp     = (const float*)d_in[4];
  const float* ffn_b  = (const float*)d_in[5];
  const float* ffn_g  = (const float*)d_in[6];
  const float* n2_b   = (const float*)d_in[7];
  const float* n2_g   = (const float*)d_in[8];
  const float* n3_b   = (const float*)d_in[9];
  const float* n3_g   = (const float*)d_in[10];
  const float* n4_b   = (const float*)d_in[11];
  const float* n4_g   = (const float*)d_in[12];
  const float* proj_b = (const float*)d_in[13];
  const float* proj_g = (const float*)d_in[14];
  SAP s1  = load_sap(d_in, 15);
  SAP s2a = load_sap(d_in, 33);
  SAP s2b = load_sap(d_in, 51);
  SAP s2c = load_sap(d_in, 69);
  SAP s3a = load_sap(d_in, 87);
  SAP s3b = load_sap(d_in, 105);
  SAP s3c = load_sap(d_in, 123);
  SAP s4a = load_sap(d_in, 141);
  SAP s4b = load_sap(d_in, 159);
  SAP s4c = load_sap(d_in, 177);
  SAP sc2 = load_sap(d_in, 195);
  SAP sc3 = load_sap(d_in, 213);
  SAP sc4 = load_sap(d_in, 231);

  // workspace carve-out (floats)
  float* W = (float*)d_ws;
  size_t off = 0;
  auto alloc = [&](long long n) {
    float* p = W + off;
    off += (size_t)((n + 255) & ~255LL);
    return p;
  };
  Scratch S;
  S.cols   = alloc(10LL * 1024 * 1024);  // max B*N*K = 9.44M (s2b)
  S.q      = alloc(4500000);             // max B*N*C = 4.19M
  S.k      = alloc(4500000);
  S.v      = alloc(4500000);
  S.scores = alloc(34LL * 1024 * 1024);  // max B*N*N = 33.6M (also FFN hidden)
  S.ctx    = alloc(4500000);
  S.t      = alloc(4500000);
  S.sum    = alloc(4096);
  S.sq     = alloc(4096);
  float* bufA = alloc(4500000);
  float* bufB = alloc(4500000);
  float* bufC = alloc(4500000);
  float* pooled = alloc(32 * 1024);
  float* pooln  = alloc(32 * 1024);

  // ---- stage 1 -------------------------------------------------------------
  sa_block(stream, s1, x, bufC, S, B, 3, 64, 64, 64, 7, 7, 2, 3);      // [32,64,32,32]

  // ---- stage 2 -------------------------------------------------------------
  sa_block(stream, s2a, bufC, bufA, S, B, 64, 32, 32, 128, 1, 1, 1, 0);  // [32,128,32,32]
  sa_block(stream, s2b, bufA, bufB, S, B, 128, 32, 32, 128, 3, 3, 2, 1); // [32,128,16,16]
  sa_block(stream, s2c, bufB, bufA, S, B, 128, 16, 16, 256, 1, 1, 1, 0); // [32,256,16,16]
  sa_block(stream, sc2, bufC, bufB, S, B, 64, 32, 32, 256, 1, 1, 2, 0);  // [32,256,16,16]
  bn_merge(stream, S, bufA, bufB, bufC, n2_g, n2_b, B, 256, 256);        // [32,256,16,16]

  // ---- stage 3 -------------------------------------------------------------
  sa_block(stream, s3a, bufC, bufA, S, B, 256, 16, 16, 256, 1, 1, 1, 0); // [32,256,16,16]
  sa_block(stream, s3b, bufA, bufB, S, B, 256, 16, 16, 256, 3, 3, 2, 1); // [32,256,8,8]
  sa_block(stream, s3c, bufB, bufA, S, B, 256, 8, 8, 512, 1, 1, 1, 0);   // [32,512,8,8]
  sa_block(stream, sc3, bufC, bufB, S, B, 256, 16, 16, 512, 1, 1, 2, 0); // [32,512,8,8]
  bn_merge(stream, S, bufA, bufB, bufC, n3_g, n3_b, B, 512, 64);         // [32,512,8,8]

  // ---- stage 4 -------------------------------------------------------------
  sa_block(stream, s4a, bufC, bufA, S, B, 512, 8, 8, 512, 1, 1, 1, 0);   // [32,512,8,8]
  sa_block(stream, s4b, bufA, bufB, S, B, 512, 8, 8, 512, 3, 3, 2, 1);   // [32,512,4,4]
  sa_block(stream, s4c, bufB, bufA, S, B, 512, 4, 4, 1024, 1, 1, 1, 0);  // [32,1024,4,4]
  sa_block(stream, sc4, bufC, bufB, S, B, 512, 8, 8, 1024, 1, 1, 2, 0);  // [32,1024,4,4]
  bn_merge(stream, S, bufA, bufB, bufC, n4_g, n4_b, B, 1024, 16);        // [32,1024,16]

  // ---- head ----------------------------------------------------------------
  // t = [B,16,1024]
  transpose_k<<<grid1d(32LL * 1024 * 16, 256), 256, 0, stream>>>(bufC, bufA, B, 1024, 16);
  // y = gelu(t @ W1) @ W2
  gemm(stream, bufA, W1, nullptr, S.scores, 512, 2048, 1024, 2048, 0, 1, 0, 0, 0);
  gelu_k<<<grid1d(512LL * 2048, 256), 256, 0, stream>>>(S.scores, 512LL * 2048);
  gemm(stream, S.scores, W2, nullptr, bufB, 512, 1024, 2048, 1024, 0, 1, 0, 0, 0);
  // t = ln(y + t)
  add_k<<<grid1d(512LL * 1024, 256), 256, 0, stream>>>(bufB, bufA, bufB, 512LL * 1024);
  layernorm_k<<<512, 256, 0, stream>>>(bufB, ffn_g, ffn_b, bufB, 512, 1024, 0);
  // mean over tokens, final LN, classifier
  mean_mid_k<<<grid1d(32LL * 1024, 256), 256, 0, stream>>>(bufB, pooled, B, 16, 1024);
  layernorm_k<<<32, 256, 0, stream>>>(pooled, proj_g, proj_b, pooln, 32, 1024, 0);
  gemm(stream, pooln, Wp, bp, (float*)d_out, 32, 1000, 1024, 1000, 0, 1, 0, 0, 0);
}
